// RotateMultiHeadAttention_154618822887
// MI455X (gfx1250) — matrix operand order repaired and checked by reading, not compiled, not run on hardware
//
#include <hip/hip_runtime.h>
#include <hip/hip_bf16.h>
#include <math.h>

// Problem constants
#define BB  4
#define SS  1024
#define HH  2048
#define NHH 16
#define DD  128

typedef __attribute__((ext_vector_type(8)))  float   v8f;
typedef __attribute__((ext_vector_type(16))) __bf16  v16bf;
typedef __attribute__((ext_vector_type(8)))  __bf16  v8bf;
typedef __attribute__((ext_vector_type(2)))  __bf16  v2bf;

// ---------------------------------------------------------------------------
// Fragment loaders (wave32 WMMA layouts, CDNA5 ISA 7.12.2)
// A operand 16x32 (MxK): lane half 0 -> K 0..7 (v0-3) and 16..23 (v4-7),
//                        lane half 1 -> K 8..15 and 24..31.
// B operand 32x16 (KxN): same K order as A (lane half 0 -> K 0..7, 16..23; half 1 -> K 8..15, 24..31).
// ---------------------------------------------------------------------------
static __device__ __forceinline__ v16bf frag_a_bf16(const __bf16* row, int k0, int kh) {
  v8bf a = *(const v8bf*)(row + k0 + kh * 8);
  v8bf b = *(const v8bf*)(row + k0 + 16 + kh * 8);
  v16bf r;
#pragma unroll
  for (int i = 0; i < 8; ++i) { r[i] = a[i]; r[i + 8] = b[i]; }
  return r;
}
static __device__ __forceinline__ v16bf frag_b_bf16(const __bf16* row, int k0, int kh) {
  v8bf a = *(const v8bf*)(row + k0 + kh * 8);
  v8bf b = *(const v8bf*)(row + k0 + 16 + kh * 8);
  v16bf r;
#pragma unroll
  for (int i = 0; i < 8; ++i) { r[i] = a[i]; r[i + 8] = b[i]; }
  return r;
}

static __device__ __forceinline__ v8f wmma_bf16(v16bf a, v16bf b, v8f c) {
  return __builtin_amdgcn_wmma_f32_16x16x32_bf16(false, a, false, b, (short)0, c, false, false);
}

static __device__ __forceinline__ float redmax16(float x) {
#pragma unroll
  for (int m = 1; m < 16; m <<= 1) x = fmaxf(x, __shfl_xor(x, m, 32));
  return x;
}
static __device__ __forceinline__ float redsum16(float x) {
#pragma unroll
  for (int m = 1; m < 16; m <<= 1) x += __shfl_xor(x, m, 32);
  return x;
}

// ---------------------------------------------------------------------------
// Elementwise fp32 -> bf16 conversion (one-time per launch; memory bound).
// n must be a multiple of 8.
// ---------------------------------------------------------------------------
__global__ void __launch_bounds__(256)
cvt_f32_bf16(const float* __restrict__ in, __bf16* __restrict__ out, int n) {
  const int i = (blockIdx.x * 256 + threadIdx.x) * 8;
  if (i < n) {
    v8f x = *(const v8f*)(in + i);
    v8bf y;
#pragma unroll
    for (int j = 0; j < 8; ++j) y[j] = (__bf16)x[j];
    *(v8bf*)(out + i) = y;
  }
}

// ---------------------------------------------------------------------------
// Kernel A: Y = X * W^T + bias, optional RoPE, write bf16 in attention layout.
// mode 0: Q -> rope, (b,h,s,d).  mode 1: K -> rope, (b,h,s,d).
// mode 2: V -> no rope, transposed (b,h,d,s).
// Block = 128 threads = 4 waves; wave w owns rows [64*bx + 16w, +16),
// columns [64*by, +64) as four 16x16 C tiles. K loop software-pipelined.
// ---------------------------------------------------------------------------
__global__ void __launch_bounds__(128)
gemm_qkv(const __bf16* __restrict__ X, const __bf16* __restrict__ W,
         const float* __restrict__ bias, const float* __restrict__ fcos,
         const float* __restrict__ fsin, __bf16* __restrict__ out, int mode) {
  const int lane = threadIdx.x & 31;
  const int wid  = threadIdx.x >> 5;
  const int kh   = lane >> 4;
  const int l15  = lane & 15;

  const int mrow  = blockIdx.x * 64 + wid * 16;
  const int ncol0 = blockIdx.y * 64;

  const __bf16* xrow = X + (size_t)(mrow + l15) * HH;
  const __bf16* wrow0 = W + (size_t)(ncol0 +  0 + l15) * HH;
  const __bf16* wrow1 = W + (size_t)(ncol0 + 16 + l15) * HH;
  const __bf16* wrow2 = W + (size_t)(ncol0 + 32 + l15) * HH;
  const __bf16* wrow3 = W + (size_t)(ncol0 + 48 + l15) * HH;

  v8f acc[4];
#pragma unroll
  for (int nt = 0; nt < 4; ++nt)
#pragma unroll
    for (int r = 0; r < 8; ++r) acc[nt][r] = 0.0f;

  // software pipeline: prefetch next k-tile before consuming current
  v16bf a_c  = frag_a_bf16(xrow, 0, kh);
  v16bf b_c0 = frag_b_bf16(wrow0, 0, kh);
  v16bf b_c1 = frag_b_bf16(wrow1, 0, kh);
  v16bf b_c2 = frag_b_bf16(wrow2, 0, kh);
  v16bf b_c3 = frag_b_bf16(wrow3, 0, kh);

  for (int k0 = 0; k0 < HH; k0 += 32) {
    const int kn = (k0 + 32) & (HH - 1);     // wraps to 0 on last iter
    v16bf a_n  = frag_a_bf16(xrow, kn, kh);
    v16bf b_n0 = frag_b_bf16(wrow0, kn, kh);
    v16bf b_n1 = frag_b_bf16(wrow1, kn, kh);
    v16bf b_n2 = frag_b_bf16(wrow2, kn, kh);
    v16bf b_n3 = frag_b_bf16(wrow3, kn, kh);
    acc[0] = wmma_bf16(a_c, b_c0, acc[0]);
    acc[1] = wmma_bf16(a_c, b_c1, acc[1]);
    acc[2] = wmma_bf16(a_c, b_c2, acc[2]);
    acc[3] = wmma_bf16(a_c, b_c3, acc[3]);
    a_c = a_n; b_c0 = b_n0; b_c1 = b_n1; b_c2 = b_n2; b_c3 = b_n3;
  }

#pragma unroll
  for (int nt = 0; nt < 4; ++nt) {
    const int col = ncol0 + nt * 16 + l15;
    const float bv = bias[col];
    const int h = col >> 7;        // col / DD
    const int d = col & (DD - 1);
    const int p = d >> 1;
    if (mode < 2) {
      // Q/K: rope, pack (even,odd) column pair -> one dword store (even lanes)
#pragma unroll
      for (int r = 0; r < 8; ++r) {
        const int row  = mrow + r + 8 * kh;
        const int bidx = row >> 10;
        const int spos = row & (SS - 1);
        float v = acc[nt][r] + bv;
        const float pv = __shfl_xor(v, 1, 32);   // pair partner (col ^ 1)
        const float c  = fcos[spos * (DD / 2) + p];
        const float s  = fsin[spos * (DD / 2) + p];
        v = (d & 1) ? fmaf(pv, s, v * c) : fmaf(-pv, s, v * c);
        const float vo = __shfl_xor(v, 1, 32);   // partner's rope result
        if (!(d & 1)) {
          v2bf t; t[0] = (__bf16)v; t[1] = (__bf16)vo;
          const size_t idx = ((size_t)(bidx * NHH + h) * SS + spos) * DD + d;
          *(v2bf*)(out + idx) = t;
        }
      }
    } else {
      // V^T: consecutive r are consecutive s -> pack pairs in-lane
#pragma unroll
      for (int r = 0; r < 8; r += 2) {
        const int row  = mrow + r + 8 * kh;
        const int bidx = row >> 10;
        const int spos = row & (SS - 1);
        v2bf t;
        t[0] = (__bf16)(acc[nt][r] + bv);
        t[1] = (__bf16)(acc[nt][r + 1] + bv);
        const size_t idx = ((size_t)(bidx * NHH + h) * DD + d) * SS + spos;
        *(v2bf*)(out + idx) = t;
      }
    }
  }
}

// ---------------------------------------------------------------------------
// Kernel B: flash attention. One wave per (b*NH+h, 16-row q tile).
// Online softmax over 32-wide K tiles; scores & P*V via bf16 WMMA.
// ---------------------------------------------------------------------------
__global__ void __launch_bounds__(128)
flash_attn(const __bf16* __restrict__ qb, const __bf16* __restrict__ kb,
           const __bf16* __restrict__ vt, const float* __restrict__ mask,
           __bf16* __restrict__ cx) {
  __shared__ __align__(16) __bf16 plds[4][16 * 32];

  const int lane = threadIdx.x & 31;
  const int wid  = threadIdx.x >> 5;
  const int kh   = lane >> 4;
  const int l15  = lane & 15;

  const int wg    = blockIdx.x * 4 + wid;
  const int bh    = wg >> 6;                 // / (SS/16)
  const int qbase = (wg & 63) * 16;
  const int bidx  = bh >> 4;                 // / NHH
  const int hidx  = bh & (NHH - 1);

  // Q fragments for this wave's 16 rows, K-dim = D = 128 -> 4 steps of 32.
  const __bf16* qrow = qb + ((size_t)bh * SS + qbase + l15) * DD;
  v16bf qf[4];
#pragma unroll
  for (int d4 = 0; d4 < 4; ++d4) qf[d4] = frag_a_bf16(qrow, d4 * 32, kh);

  float mstate[8], lstate[8];
  v8f o[8];
#pragma unroll
  for (int r = 0; r < 8; ++r) { mstate[r] = -3.0e38f; lstate[r] = 0.0f; }
#pragma unroll
  for (int dt = 0; dt < 8; ++dt)
#pragma unroll
    for (int r = 0; r < 8; ++r) o[dt][r] = 0.0f;

  const float scale = 0.022097086912079608f;  // 1/sqrt(H) = 1/sqrt(2048)

  for (int kt = 0; kt < SS; kt += 32) {
    // --- scores: two 16x16 tiles (k columns kt..kt+15, kt+16..kt+31) ---
    v8f s0, s1;
#pragma unroll
    for (int r = 0; r < 8; ++r) { s0[r] = 0.0f; s1[r] = 0.0f; }
    const __bf16* krow0 = kb + ((size_t)bh * SS + kt + l15) * DD;
    const __bf16* krow1 = krow0 + (size_t)16 * DD;
#pragma unroll
    for (int d4 = 0; d4 < 4; ++d4) {
      v16bf b0 = frag_b_bf16(krow0, d4 * 32, kh);
      v16bf b1 = frag_b_bf16(krow1, d4 * 32, kh);
      s0 = wmma_bf16(qf[d4], b0, s0);
      s1 = wmma_bf16(qf[d4], b1, s1);
    }
    const float mk0 = mask[(size_t)bidx * SS + kt + l15];
    const float mk1 = mask[(size_t)bidx * SS + kt + 16 + l15];
    const float add0 = (mk0 == 0.0f) ? -1.0e9f : 0.0f;
    const float add1 = (mk1 == 0.0f) ? -1.0e9f : 0.0f;

    // --- online softmax update ---
    float p0[8], p1[8], al[8];
#pragma unroll
    for (int r = 0; r < 8; ++r) {
      const float a0 = fmaf(s0[r], scale, add0);
      const float a1 = fmaf(s1[r], scale, add1);
      const float mt = redmax16(fmaxf(a0, a1));
      const float mn = fmaxf(mstate[r], mt);
      al[r] = __expf(mstate[r] - mn);
      p0[r] = __expf(a0 - mn);
      p1[r] = __expf(a1 - mn);
      const float rs = redsum16(p0[r] + p1[r]);
      lstate[r] = lstate[r] * al[r] + rs;
      mstate[r] = mn;
    }
#pragma unroll
    for (int dt = 0; dt < 8; ++dt)
#pragma unroll
      for (int r = 0; r < 8; ++r) o[dt][r] *= al[r];

    // --- re-shape P (C layout) -> A fragment via LDS ---
    __bf16* pl = plds[wid];
#pragma unroll
    for (int r = 0; r < 8; ++r) {
      const int row = r + 8 * kh;
      pl[row * 32 + l15]      = (__bf16)p0[r];
      pl[row * 32 + 16 + l15] = (__bf16)p1[r];
    }
    asm volatile("s_wait_dscnt 0" ::: "memory");
    v16bf pf = frag_a_bf16(pl + l15 * 32, 0, kh);

    // --- O += P * V  (V stored transposed: rows are d, contiguous in k) ---
#pragma unroll
    for (int dt = 0; dt < 8; ++dt) {
      const __bf16* vrow = vt + ((size_t)bh * DD + dt * 16 + l15) * SS;
      v16bf vf = frag_b_bf16(vrow, kt, kh);
      o[dt] = wmma_bf16(pf, vf, o[dt]);
    }
  }

  // --- epilogue: normalize and write ctx (b, s, h*D+d), packed dword stores ---
  float inv[8];
#pragma unroll
  for (int r = 0; r < 8; ++r) inv[r] = 1.0f / lstate[r];
#pragma unroll
  for (int dt = 0; dt < 8; ++dt) {
    const int col = hidx * DD + dt * 16 + l15;
#pragma unroll
    for (int r = 0; r < 8; ++r) {
      const int srow = qbase + r + 8 * kh;
      const float v  = o[dt][r] * inv[r];
      const float vo = __shfl_xor(v, 1, 32);    // partner column (col^1)
      if (!(l15 & 1)) {
        v2bf t; t[0] = (__bf16)v; t[1] = (__bf16)vo;
        *(v2bf*)(cx + ((size_t)bidx * SS + srow) * HH + col) = t;
      }
    }
  }
}

// ---------------------------------------------------------------------------
// Kernel C: out = ctx(bf16) * Wo^T + bo, fp32 result. Pipelined like kernel A.
// ---------------------------------------------------------------------------
__global__ void __launch_bounds__(128)
gemm_out(const __bf16* __restrict__ Xc, const __bf16* __restrict__ W,
         const float* __restrict__ bias, float* __restrict__ out) {
  const int lane = threadIdx.x & 31;
  const int wid  = threadIdx.x >> 5;
  const int kh   = lane >> 4;
  const int l15  = lane & 15;

  const int mrow  = blockIdx.x * 64 + wid * 16;
  const int ncol0 = blockIdx.y * 64;

  const __bf16* xrow = Xc + (size_t)(mrow + l15) * HH;
  const __bf16* wrow0 = W + (size_t)(ncol0 +  0 + l15) * HH;
  const __bf16* wrow1 = W + (size_t)(ncol0 + 16 + l15) * HH;
  const __bf16* wrow2 = W + (size_t)(ncol0 + 32 + l15) * HH;
  const __bf16* wrow3 = W + (size_t)(ncol0 + 48 + l15) * HH;

  v8f acc[4];
#pragma unroll
  for (int nt = 0; nt < 4; ++nt)
#pragma unroll
    for (int r = 0; r < 8; ++r) acc[nt][r] = 0.0f;

  v16bf a_c  = frag_a_bf16(xrow, 0, kh);
  v16bf b_c0 = frag_b_bf16(wrow0, 0, kh);
  v16bf b_c1 = frag_b_bf16(wrow1, 0, kh);
  v16bf b_c2 = frag_b_bf16(wrow2, 0, kh);
  v16bf b_c3 = frag_b_bf16(wrow3, 0, kh);

  for (int k0 = 0; k0 < HH; k0 += 32) {
    const int kn = (k0 + 32) & (HH - 1);
    v16bf a_n  = frag_a_bf16(xrow, kn, kh);
    v16bf b_n0 = frag_b_bf16(wrow0, kn, kh);
    v16bf b_n1 = frag_b_bf16(wrow1, kn, kh);
    v16bf b_n2 = frag_b_bf16(wrow2, kn, kh);
    v16bf b_n3 = frag_b_bf16(wrow3, kn, kh);
    acc[0] = wmma_bf16(a_c, b_c0, acc[0]);
    acc[1] = wmma_bf16(a_c, b_c1, acc[1]);
    acc[2] = wmma_bf16(a_c, b_c2, acc[2]);
    acc[3] = wmma_bf16(a_c, b_c3, acc[3]);
    a_c = a_n; b_c0 = b_n0; b_c1 = b_n1; b_c2 = b_n2; b_c3 = b_n3;
  }

#pragma unroll
  for (int nt = 0; nt < 4; ++nt) {
    const int col = ncol0 + nt * 16 + l15;
    const float bv = bias[col];
#pragma unroll
    for (int r = 0; r < 8; ++r) {
      const int row = mrow + r + 8 * kh;
      out[(size_t)row * HH + col] = acc[nt][r] + bv;
    }
  }
}

// ---------------------------------------------------------------------------
extern "C" void kernel_launch(void* const* d_in, const int* in_sizes, int n_in,
                              void* d_out, int out_size, void* d_ws, size_t ws_size,
                              hipStream_t stream) {
  const float* hs   = (const float*)d_in[0];
  const float* fcos = (const float*)d_in[1];
  const float* fsin = (const float*)d_in[2];
  const float* mask = (const float*)d_in[3];
  const float* Wq   = (const float*)d_in[4];
  const float* bq   = (const float*)d_in[5];
  const float* Wk   = (const float*)d_in[6];
  const float* bk   = (const float*)d_in[7];
  const float* Wv   = (const float*)d_in[8];
  const float* bv   = (const float*)d_in[9];
  const float* Wo   = (const float*)d_in[10];
  const float* bo   = (const float*)d_in[11];
  float* out = (float*)d_out;

  const size_t act  = (size_t)BB * SS * HH;   // 8M elements
  const size_t wel  = (size_t)HH * HH;        // 4M elements
  char* ws = (char*)d_ws;
  __bf16* qb  = (__bf16*)(ws);                         // 16 MB
  __bf16* kb  = (__bf16*)(ws + act * 2);               // 16 MB
  __bf16* vtb = (__bf16*)(ws + act * 4);               // 16 MB
  __bf16* cxb = (__bf16*)(ws + act * 6);               // 16 MB
  __bf16* hsb = (__bf16*)(ws + act * 8);               // 16 MB
  __bf16* wqb = (__bf16*)(ws + act * 10);              // 8 MB
  __bf16* wkb = (__bf16*)(ws + act * 10 + wel * 2);    // 8 MB
  __bf16* wvb = (__bf16*)(ws + act * 10 + wel * 4);    // 8 MB
  __bf16* wob = (__bf16*)(ws + act * 10 + wel * 6);    // 8 MB

  // one-time fp32 -> bf16 conversions (memory bound)
  cvt_f32_bf16<<<dim3((int)(act / 2048)), dim3(256), 0, stream>>>(hs, hsb, (int)act);
  cvt_f32_bf16<<<dim3((int)(wel / 2048)), dim3(256), 0, stream>>>(Wq, wqb, (int)wel);
  cvt_f32_bf16<<<dim3((int)(wel / 2048)), dim3(256), 0, stream>>>(Wk, wkb, (int)wel);
  cvt_f32_bf16<<<dim3((int)(wel / 2048)), dim3(256), 0, stream>>>(Wv, wvb, (int)wel);
  cvt_f32_bf16<<<dim3((int)(wel / 2048)), dim3(256), 0, stream>>>(Wo, wob, (int)wel);

  dim3 blk(128);
  dim3 gg((BB * SS) / 64, HH / 64);   // (64, 32)

  gemm_qkv<<<gg, blk, 0, stream>>>(hsb, wqb, bq, fcos, fsin, qb, 0);
  gemm_qkv<<<gg, blk, 0, stream>>>(hsb, wkb, bk, fcos, fsin, kb, 1);
  gemm_qkv<<<gg, blk, 0, stream>>>(hsb, wvb, bv, fcos, fsin, vtb, 2);

  dim3 ga((BB * NHH * (SS / 16)) / 4);  // 1024 blocks of 4 waves
  flash_attn<<<ga, blk, 0, stream>>>(qb, kb, vtb, mask, cxb);

  gemm_out<<<gg, blk, 0, stream>>>(cxb, wob, bo, out);
}
